// DeepFM_MTL_87995289960919
// MI455X (gfx1250) — matrix operand order repaired and checked by reading, not compiled, not run on hardware
//
#include <hip/hip_runtime.h>
#include <hip/hip_bf16.h>

typedef __attribute__((ext_vector_type(16))) _Float16 v16h;
typedef __attribute__((ext_vector_type(8)))  _Float16 v8h;
typedef __attribute__((ext_vector_type(2)))  _Float16 v2h;
typedef __attribute__((ext_vector_type(8)))  float    v8f;

#define Bn    16384
#define En    16
#define LSn   50
#define LHn   200
#define KPAD  224      // padded K for all GEMM layers (multiple of 32)
#define NOUT  200
#define NTILE 13       // ceil(200/16)
#define KTILE 7        // 224/32
#define PWELEMS (KTILE*NTILE*32*16)   // 46592 halves per layer (93,184 B)

__constant__ int d_OFF[8] = {0, 1000000, 1001000, 2001000,
                             3001000, 3002000, 3002004, 3102004};
#define ITEM_ID_OFF   1001000
#define ITEM_CITY_OFF 3001000

__device__ __forceinline__ float red4cls(float x) {   // sum over row-class groups
  x += __shfl_xor(x, 8, 32);
  x += __shfl_xor(x, 16, 32);
  return x;
}

// ---------------------------------------------------------------------------
// Pack W1/W2/W3 (f32 row-major [K,200]) into WMMA B-fragment layout, f16:
// pw[((kt*NTILE+nt)*32 + lane)*16 + h] = W[kt*32 + (lane>>4)*16 + h][nt*16 + (lane&15)]
// ---------------------------------------------------------------------------
__global__ void pack_weights(const float* __restrict__ W1,
                             const float* __restrict__ W2,
                             const float* __restrict__ W3,
                             _Float16* __restrict__ pw) {
  int tid = blockIdx.x * blockDim.x + threadIdx.x;
  if (tid >= 3 * PWELEMS) return;
  int layer = tid / PWELEMS;
  int t = tid % PWELEMS;
  int kt = t / (NTILE * 512);
  int r  = t % (NTILE * 512);
  int nt = r / 512;
  int q  = r % 512;
  int lane = q >> 4;
  int h    = q & 15;
  int n = nt * 16 + (lane & 15);
  int k = kt * 32 + (lane >> 4) * 16 + h;
  const float* W = (layer == 0) ? W1 : (layer == 1 ? W2 : W3);
  int kact = (layer == 0) ? 194 : 200;
  float v = (k < kact && n < NOUT) ? W[k * NOUT + n] : 0.0f;
  pw[tid] = (_Float16)v;
}

// ---------------------------------------------------------------------------
// One wave32 per sample.  b64 (float2) gathers: 8 lanes per row, 4 rows per
// instruction.  lane -> element pair p = lane&7 (elements 2p,2p+1), row class
// c = lane>>3.  Class partials combined with shfl_xor(8) + shfl_xor(16).
// ---------------------------------------------------------------------------
__global__ void embed_fuse(const int* __restrict__ sparse, const float* __restrict__ dense,
                           const int* __restrict__ seqA,   const int* __restrict__ seqG,
                           const int* __restrict__ histI,  const int* __restrict__ histC,
                           const float* __restrict__ linW, const float* __restrict__ linB,
                           const float* __restrict__ linEmb, const float* __restrict__ fmEmb,
                           const float* __restrict__ embA, const float* __restrict__ embG,
                           _Float16* __restrict__ act0, float* __restrict__ fo_so) {
  int wave = (int)((blockIdx.x * blockDim.x + threadIdx.x) >> 5);
  int lane = threadIdx.x & 31;
  if (wave >= Bn) return;
  int p = lane & 7;          // element pair: elements 2p, 2p+1
  int c = lane >> 3;         // row class 0..3

  // ---- pooled sequences: mean over rows, 4 rows per iteration ----
  float pax = 0.f, pay = 0.f, pgx = 0.f, pgy = 0.f;
  for (int i = 0; i < LSn / 4; ++i) {          // 48 rows
    int t = 4 * i + c;
    const float2 a = *(const float2*)(embA + (size_t)seqA[wave * LSn + t] * En + 2 * p);
    const float2 g = *(const float2*)(embG + (size_t)seqG[wave * LSn + t] * En + 2 * p);
    pax += a.x; pay += a.y; pgx += g.x; pgy += g.y;
  }
  if (c < 2) {                                  // tail rows 48,49
    int t = 48 + c;
    const float2 a = *(const float2*)(embA + (size_t)seqA[wave * LSn + t] * En + 2 * p);
    const float2 g = *(const float2*)(embG + (size_t)seqG[wave * LSn + t] * En + 2 * p);
    pax += a.x; pay += a.y; pgx += g.x; pgy += g.y;
  }
  float pix = 0.f, piy = 0.f, pcx = 0.f, pcy = 0.f;
  for (int i = 0; i < LHn / 4; ++i) {          // 200 rows
    int t = 4 * i + c;
    const float2 a = *(const float2*)(fmEmb + ((size_t)histI[wave * LHn + t] + ITEM_ID_OFF)  * En + 2 * p);
    const float2 g = *(const float2*)(fmEmb + ((size_t)histC[wave * LHn + t] + ITEM_CITY_OFF) * En + 2 * p);
    pix += a.x; piy += a.y; pcx += g.x; pcy += g.y;
  }
  pax = red4cls(pax) * (1.0f / LSn);  pay = red4cls(pay) * (1.0f / LSn);
  pgx = red4cls(pgx) * (1.0f / LSn);  pgy = red4cls(pgy) * (1.0f / LSn);
  pix = red4cls(pix) * (1.0f / LHn);  piy = red4cls(piy) * (1.0f / LHn);
  pcx = red4cls(pcx) * (1.0f / LHn);  pcy = red4cls(pcy) * (1.0f / LHn);

  // ---- 8 sparse fm embeddings: 4 rows per iteration ----
  float fsx = 0.f, fsy = 0.f, fqx = 0.f, fqy = 0.f;
  size_t abase = (size_t)wave * KPAD;
  #pragma unroll
  for (int i = 0; i < 2; ++i) {
    int j = 4 * i + c;                          // 0..7
    int idx = sparse[wave * 8 + j] + d_OFF[j];
    const float2 x = *(const float2*)(fmEmb + (size_t)idx * En + 2 * p);
    fsx += x.x; fsy += x.y; fqx += x.x * x.x; fqy += x.y * x.y;
    v2h hv; hv.x = (_Float16)x.x; hv.y = (_Float16)x.y;
    *(v2h*)(act0 + abase + 2 + j * En + 2 * p) = hv;   // 4B-aligned packed store
  }

  // ---- FM second order ----
  float sum0 = red4cls(fsx) + pax + pgx + pix + pcx;
  float sum1 = red4cls(fsy) + pay + pgy + piy + pcy;
  float sq0  = red4cls(fqx) + pax*pax + pgx*pgx + pix*pix + pcx*pcx;
  float sq1  = red4cls(fqy) + pay*pay + pgy*pgy + piy*piy + pcy*pcy;
  float term = (sum0 * sum0 - sq0) + (sum1 * sum1 - sq1);
  term += __shfl_xor(term, 4, 32);
  term += __shfl_xor(term, 2, 32);
  term += __shfl_xor(term, 1, 32);
  float second = 0.5f * term;                   // replicated in each 8-lane group

  // ---- write pooled vectors (class 0 lanes) + zero K-padding ----
  if (c == 0) {
    v2h h;
    h.x = (_Float16)pax; h.y = (_Float16)pay; *(v2h*)(act0 + abase + 130 + 2 * p) = h;
    h.x = (_Float16)pgx; h.y = (_Float16)pgy; *(v2h*)(act0 + abase + 146 + 2 * p) = h;
    h.x = (_Float16)pix; h.y = (_Float16)piy; *(v2h*)(act0 + abase + 162 + 2 * p) = h;
    h.x = (_Float16)pcx; h.y = (_Float16)pcy; *(v2h*)(act0 + abase + 178 + 2 * p) = h;
  }
  if (lane < 30) act0[abase + 194 + lane] = (_Float16)0.0f;

  // ---- first order (linear sparse on lanes 0-7, dense on lane 0) ----
  float lv = 0.f;
  if (lane < 8) lv = linEmb[(size_t)sparse[wave * 8 + lane] + d_OFF[lane]];
  lv += __shfl_xor(lv, 1, 32);
  lv += __shfl_xor(lv, 2, 32);
  lv += __shfl_xor(lv, 4, 32);

  if (lane == 0) {
    float d0 = dense[wave * 2 + 0];
    float d1 = dense[wave * 2 + 1];
    act0[abase + 0] = (_Float16)d0;
    act0[abase + 1] = (_Float16)d1;
    float first = d0 * linW[0] + d1 * linW[1] + linB[0] + lv;
    fo_so[wave] = first + second;
  }
}

// ---------------------------------------------------------------------------
// GEMM + bias + ReLU via V_WMMA_F32_16X16X32_F16.
// Whole layer's packed weights (93 KB) staged into LDS with
// GLOBAL_LOAD_ASYNC_TO_LDS_B128 (ASYNCcnt), shared by the block's 8 waves.
// One wave owns a 16-row M block and all 13 N-tiles (13 live v8f accs);
// kt outer / nt inner: A loaded once per kt, 13 independent WMMAs fed from DS.
// ---------------------------------------------------------------------------
__global__ void mlp_gemm(const _Float16* __restrict__ A,
                         const _Float16* __restrict__ PW,
                         const float* __restrict__ bias,
                         _Float16* __restrict__ Out) {
  extern __shared__ _Float16 sB[];              // PWELEMS halves

  // ---- async stage PW -> LDS in 16B chunks ----
  for (int chunk = threadIdx.x; chunk < PWELEMS / 8; chunk += 256) {
    unsigned lds_off = (unsigned)(uintptr_t)(&sB[(size_t)chunk * 8]);
    unsigned long long gaddr = (unsigned long long)(uintptr_t)(PW + (size_t)chunk * 8);
    asm volatile("global_load_async_to_lds_b128 %0, %1, off"
                 :: "v"(lds_off), "v"(gaddr) : "memory");
  }
  asm volatile("s_wait_asynccnt 0x0" ::: "memory");
  __syncthreads();

  int wave = (int)((blockIdx.x * blockDim.x + threadIdx.x) >> 5);
  int lane = threadIdx.x & 31;
  if (wave >= Bn / 16) return;
  int mt = wave;

  v8f acc[NTILE];
  #pragma unroll
  for (int nt = 0; nt < NTILE; ++nt) acc[nt] = (v8f){};

  // A fragment addressing: lane -> row m = mt*16 + (lane&15),
// K halves {base..base+15}, base = (lane>>4)*16
  const _Float16* arow = A + (size_t)(mt * 16 + (lane & 15)) * KPAD + ((lane >> 4) * 16);

  #pragma unroll
  for (int kt = 0; kt < KTILE; ++kt) {
    v8h lo = *(const v8h*)(arow);
    v8h hi = *(const v8h*)(arow + 8);
    v16h a = __builtin_shufflevector(lo, hi, 0,1,2,3,4,5,6,7,8,9,10,11,12,13,14,15);
    #pragma unroll
    for (int nt = 0; nt < NTILE; ++nt) {
      v16h bfrag = *(const v16h*)(&sB[(size_t)(kt * NTILE + nt) * 512 + (size_t)lane * 16]);
      acc[nt] = __builtin_amdgcn_wmma_f32_16x16x32_f16(
          false, a, false, bfrag, (short)0, acc[nt], false, false);
    }
    arow += 32;
  }

  int col = lane & 15;
  int mbase = mt * 16 + ((lane >> 4) << 3);    // C/D layout: lanes 16-31 -> rows +8
  #pragma unroll
  for (int nt = 0; nt < NTILE; ++nt) {
    int n = nt * 16 + col;
    float bn = (n < NOUT) ? bias[n] : 0.0f;
    #pragma unroll
    for (int r2 = 0; r2 < 8; ++r2) {
      float v = acc[nt][r2] + bn;
      v = v > 0.0f ? v : 0.0f;                 // ReLU (all three layers)
      _Float16 hv = (n < NOUT) ? (_Float16)v : (_Float16)0.0f;
      Out[(size_t)(mbase + r2) * KPAD + n] = hv;
    }
  }
  #pragma unroll
  for (int r2 = 0; r2 < 8; ++r2)               // cols 208..223 never touched by tiles
    Out[(size_t)(mbase + r2) * KPAD + 208 + col] = (_Float16)0.0f;
}

// ---------------------------------------------------------------------------
// Head: one wave per sample, coalesced K-strided dot + xor reduction.
// ---------------------------------------------------------------------------
__global__ void head_kernel(const _Float16* __restrict__ h3, const float* __restrict__ fo_so,
                            const float* __restrict__ W4, const float* __restrict__ b4,
                            const float* __restrict__ Wf, const float* __restrict__ bf,
                            const float* __restrict__ Wl, const float* __restrict__ bl,
                            float* __restrict__ out) {
  int wave = (int)((blockIdx.x * blockDim.x + threadIdx.x) >> 5);
  int lane = threadIdx.x & 31;
  if (wave >= Bn) return;
  const _Float16* row = h3 + (size_t)wave * KPAD;
  float acc = 0.f;
  for (int k = lane; k < NOUT; k += 32)
    acc += (float)row[k] * W4[k];
  acc += __shfl_xor(acc, 16, 32);
  acc += __shfl_xor(acc, 8, 32);
  acc += __shfl_xor(acc, 4, 32);
  acc += __shfl_xor(acc, 2, 32);
  acc += __shfl_xor(acc, 1, 32);
  if (lane == 0) {
    float logits = fo_so[wave] + acc + b4[0];
    out[wave]      = 1.0f / (1.0f + __expf(-(logits * Wf[0] + bf[0])));
    out[Bn + wave] = 1.0f / (1.0f + __expf(-(logits * Wl[0] + bl[0])));
  }
}

// ---------------------------------------------------------------------------
extern "C" void kernel_launch(void* const* d_in, const int* in_sizes, int n_in,
                              void* d_out, int out_size, void* d_ws, size_t ws_size,
                              hipStream_t stream) {
  (void)in_sizes; (void)n_in; (void)out_size; (void)ws_size;

  const int*   sparse = (const int*)  d_in[0];
  const float* dense  = (const float*)d_in[1];
  const int*   seqA   = (const int*)  d_in[2];
  const int*   seqG   = (const int*)  d_in[3];
  const int*   histI  = (const int*)  d_in[4];
  const int*   histC  = (const int*)  d_in[5];
  const float* linW   = (const float*)d_in[6];
  const float* linB   = (const float*)d_in[7];
  const float* linEmb = (const float*)d_in[8];
  const float* fmEmb  = (const float*)d_in[9];
  const float* embA   = (const float*)d_in[10];
  const float* embG   = (const float*)d_in[11];
  const float* W1 = (const float*)d_in[12];
  const float* b1 = (const float*)d_in[13];
  const float* W2 = (const float*)d_in[14];
  const float* b2 = (const float*)d_in[15];
  const float* W3 = (const float*)d_in[16];
  const float* b3 = (const float*)d_in[17];
  const float* W4 = (const float*)d_in[18];
  const float* b4 = (const float*)d_in[19];
  const float* Wf = (const float*)d_in[20];
  const float* bf = (const float*)d_in[21];
  const float* Wl = (const float*)d_in[22];
  const float* bl = (const float*)d_in[23];

  // workspace layout (all fully re-initialized each launch)
  _Float16* act0  = (_Float16*)d_ws;                       // B*KPAD f16
  _Float16* act1  = act0 + (size_t)Bn * KPAD;              // B*KPAD f16
  float*    fo_so = (float*)(act1 + (size_t)Bn * KPAD);    // B f32
  _Float16* pw    = (_Float16*)(fo_so + Bn);               // 3 * PWELEMS f16
  _Float16* pw1 = pw;
  _Float16* pw2 = pw + PWELEMS;
  _Float16* pw3 = pw + 2 * (size_t)PWELEMS;

  pack_weights<<<(3 * PWELEMS + 255) / 256, 256, 0, stream>>>(W1, W2, W3, pw);

  embed_fuse<<<(Bn * 32 + 255) / 256, 256, 0, stream>>>(
      sparse, dense, seqA, seqG, histI, histC,
      linW, linB, linEmb, fmEmb, embA, embG, act0, fo_so);

  const int gemm_waves  = Bn / 16;                          // 1024 waves
  const int gemm_blocks = (gemm_waves * 32 + 255) / 256;    // 128 blocks
  const size_t ldsB = (size_t)PWELEMS * sizeof(_Float16);   // 93,184 B
  mlp_gemm<<<gemm_blocks, 256, ldsB, stream>>>(act0, pw1, b1, act1);  // layer 1
  mlp_gemm<<<gemm_blocks, 256, ldsB, stream>>>(act1, pw2, b2, act0);  // layer 2
  mlp_gemm<<<gemm_blocks, 256, ldsB, stream>>>(act0, pw3, b3, act1);  // layer 3

  head_kernel<<<(Bn * 32 + 255) / 256, 256, 0, stream>>>(
      act1, fo_so, W4, b4, Wf, bf, Wl, bl, (float*)d_out);
}
